// Dotproduct_scaled_91207925498491
// MI455X (gfx1250) — matrix operand order repaired and checked by reading, not compiled, not run on hardware
//
#include <hip/hip_runtime.h>

typedef _Float16 v8h  __attribute__((ext_vector_type(8)));
typedef _Float16 v16h __attribute__((ext_vector_type(16)));
typedef float    v4f  __attribute__((ext_vector_type(4)));
typedef float    v8f  __attribute__((ext_vector_type(8)));

#define DIM        512
#define SCALE      0.04419417382415922f   // 1/sqrt(512)
#define LN_EPS     1e-3f

// ws layout (in _Float16 elements)
#define QH_OFF     0u
#define KH_OFF     16777216u   // 32768*512
#define VT_OFF     33554432u
#define WT_OFF     50331648u   // 3 transposed f16 weight matrices (786432)
#define XH_OFF     51118080u   // X converted to f16 (16777216)

__device__ __forceinline__ v8h ld8h(const _Float16* p) {
    return *reinterpret_cast<const v8h*>(p);
}
__device__ __forceinline__ v16h cat16(v8h lo, v8h hi) {
    return __builtin_shufflevector(lo, hi, 0,1,2,3,4,5,6,7,8,9,10,11,12,13,14,15);
}
__device__ __forceinline__ v16h ld16h(const _Float16* p) {
    return cat16(ld8h(p), ld8h(p + 8));
}
__device__ __forceinline__ v8f wmma16(v16h a, v16h b, v8f c) {
    return __builtin_amdgcn_wmma_f32_16x16x32_f16(false, a, false, b, (short)0, c, false, false);
}

// ---------------------------------------------------------------------------
// Prep 1: weights -> f16 transposed.  Wt[which][n][k] = dw[which][k][n]
// ---------------------------------------------------------------------------
__global__ __launch_bounds__(256) void prep_weights(
    const float* __restrict__ dw1, const float* __restrict__ dw2,
    const float* __restrict__ dw3, _Float16* __restrict__ Wt)
{
    unsigned i = blockIdx.x * 256u + threadIdx.x;   // 3*512*512 total
    if (i >= 3u * DIM * DIM) return;
    unsigned which = i / (DIM * DIM);
    unsigned r = i % (DIM * DIM);
    unsigned n = r / DIM, k = r % DIM;
    const float* w = (which == 0) ? dw1 : (which == 1) ? dw2 : dw3;
    Wt[i] = (_Float16)w[k * DIM + n];
}

// ---------------------------------------------------------------------------
// Prep 2: X -> f16, straight layout.  8 elements per thread, vectorized.
// ---------------------------------------------------------------------------
__global__ __launch_bounds__(256) void prep_xh(
    const float* __restrict__ X, _Float16* __restrict__ Xh)
{
    size_t i = ((size_t)blockIdx.x * 256 + threadIdx.x) * 8;   // grid sized exactly
    v4f a = *reinterpret_cast<const v4f*>(X + i);
    v4f b = *reinterpret_cast<const v4f*>(X + i + 4);
    v8h o;
#pragma unroll
    for (int j = 0; j < 4; ++j) { o[j] = (_Float16)a[j]; o[4 + j] = (_Float16)b[j]; }
    *reinterpret_cast<v8h*>(Xh + i) = o;
}

// ---------------------------------------------------------------------------
// QKV GEMM: each wave computes a 16(M) x 64(N) strip (4 accumulators).
// Ping-pong buffers alternate structurally (unroll-by-2): no register copies,
// prefetch of step k+1 stays in flight under step-k WMMAs.
// ---------------------------------------------------------------------------
__global__ __launch_bounds__(256) void qkv_gemm(
    const _Float16* __restrict__ Xh,
    const float* __restrict__ db1, const float* __restrict__ db2,
    const float* __restrict__ db3,
    const _Float16* __restrict__ Wt,
    _Float16* __restrict__ Qh, _Float16* __restrict__ Kh,
    _Float16* __restrict__ Vt)
{
    const int lane = threadIdx.x & 31;
    const int wave = threadIdx.x >> 5;
    const int l15  = lane & 15;
    const int hi   = lane >> 4;

    int t = blockIdx.x * 8 + wave;          // strip id, < 3*2048*8
    int which = t / 16384;                  // 0=Q 1=K 2=V
    int rem   = t % 16384;
    int tm    = rem >> 3;                   // M tile (0..2047)
    int ng    = rem & 7;                    // N group of 64 (0..7)

    const _Float16* xrow = Xh + (size_t)(tm * 16 + l15) * DIM + hi * 16;
    const _Float16* w0   = Wt + (size_t)which * (DIM * DIM)
                              + (size_t)(ng * 64 + l15) * DIM + hi * 16;

    auto ldA = [&](int kb) { return cat16(ld8h(xrow + kb), ld8h(xrow + kb + 8)); };
    auto ldB = [&](int kb, v16h* bb) {
#pragma unroll
        for (int u = 0; u < 4; ++u) bb[u] = ld16h(w0 + u * (16 * DIM) + kb);
    };

    v8f acc[4] = {};
    v16h a0, a1, b0[4], b1[4];
    a0 = ldA(0); ldB(0, b0);

#pragma unroll 1
    for (int kb = 0; kb < DIM - 64; kb += 64) {
        a1 = ldA(kb + 32); ldB(kb + 32, b1);
#pragma unroll
        for (int u = 0; u < 4; ++u) acc[u] = wmma16(a0, b0[u], acc[u]);
        a0 = ldA(kb + 64); ldB(kb + 64, b0);
#pragma unroll
        for (int u = 0; u < 4; ++u) acc[u] = wmma16(a1, b1[u], acc[u]);
    }
    a1 = ldA(DIM - 32); ldB(DIM - 32, b1);
#pragma unroll
    for (int u = 0; u < 4; ++u) acc[u] = wmma16(a0, b0[u], acc[u]);
#pragma unroll
    for (int u = 0; u < 4; ++u) acc[u] = wmma16(a1, b1[u], acc[u]);

    const float* db = (which == 0) ? db1 : (which == 1) ? db2 : db3;

#pragma unroll
    for (int u = 0; u < 4; ++u) {
        int ncol = ng * 64 + u * 16 + l15;
        float bias = db[ncol];
        if (which < 2) {
            _Float16* dst = which ? Kh : Qh;    // row-major [32768][512]
#pragma unroll
            for (int v = 0; v < 8; ++v) {
                int m = tm * 16 + v + 8 * hi;
                dst[(size_t)m * DIM + ncol] = (_Float16)(acc[u][v] + bias);
            }
        } else {
            // transposed store: Vt[(blk*512 + d)*512 + q], 8 consecutive q/lane
            int blk = (tm * 16) >> 9;
            int qb  = ((tm * 16) & 511) + 8 * hi;
            v8h pk;
#pragma unroll
            for (int v = 0; v < 8; ++v) pk[v] = (_Float16)(acc[u][v] + bias);
            *reinterpret_cast<v8h*>(Vt + ((size_t)blk * DIM + ncol) * DIM + qb) = pk;
        }
    }
}

// ---------------------------------------------------------------------------
// Fused: scores -> mask -> softmax -> P@V -> residual -> LayerNorm
//   One workgroup (512 thr = 16 waves) per (block, 64-row q chunk).
//   Wave w: q-tile = w&3, 8 k-tiles (or d-tiles) = (w>>2)*8 .. +7
// ---------------------------------------------------------------------------
__global__ __launch_bounds__(512) void attn_fused(
    const float* __restrict__ X, const float* __restrict__ mask,
    const _Float16* __restrict__ Qh, const _Float16* __restrict__ Kh,
    const _Float16* __restrict__ Vt, float* __restrict__ out)
{
    __shared__ float    s[64 * 513];        // scores / exp / out staging (~128KB)
    __shared__ _Float16 Ph[64 * 520];       // normalized P, f16 (~65KB)
    __shared__ float    red[64 * 8];
    __shared__ float    red2[64 * 8];
    __shared__ float    rMean[64];
    __shared__ float    rRstd[64];

    const int tid   = threadIdx.x;
    const int blk   = blockIdx.x >> 3;      // 0..63
    const int chunk = blockIdx.x & 7;       // 0..7
    const int lane  = tid & 31;
    const int wave  = tid >> 5;             // 0..15
    const int l15   = lane & 15;
    const int hi    = lane >> 4;
    const int qg0   = blk * DIM + chunk * 64;   // global row base of this chunk
    const int qt    = wave & 3;             // q-tile 0..3
    const int grp   = wave >> 2;            // tile-group 0..3 (of 8 tiles)

    // ---- scores: ping-pong A/B buffers, 8 WMMAs per k-step -----------------
    {
        const _Float16* qp  = Qh + (size_t)(qg0 + qt * 16 + l15) * DIM + hi * 16;
        const _Float16* kp0 = Kh + ((size_t)blk * DIM + grp * 128 + l15) * DIM + hi * 16;
        auto ldA = [&](int d) { return cat16(ld8h(qp + d), ld8h(qp + d + 8)); };
        auto ldB = [&](int d, v16h* bb) {
#pragma unroll
            for (int jj = 0; jj < 8; ++jj) bb[jj] = ld16h(kp0 + jj * (16 * DIM) + d);
        };

        v8f acc[8] = {};
        v16h a0, a1, b0[8], b1[8];
        a0 = ldA(0); ldB(0, b0);

#pragma unroll 1
        for (int d = 0; d < DIM - 64; d += 64) {
            a1 = ldA(d + 32); ldB(d + 32, b1);
#pragma unroll
            for (int jj = 0; jj < 8; ++jj) acc[jj] = wmma16(a0, b0[jj], acc[jj]);
            a0 = ldA(d + 64); ldB(d + 64, b0);
#pragma unroll
            for (int jj = 0; jj < 8; ++jj) acc[jj] = wmma16(a1, b1[jj], acc[jj]);
        }
        a1 = ldA(DIM - 32); ldB(DIM - 32, b1);
#pragma unroll
        for (int jj = 0; jj < 8; ++jj) acc[jj] = wmma16(a0, b0[jj], acc[jj]);
#pragma unroll
        for (int jj = 0; jj < 8; ++jj) acc[jj] = wmma16(a1, b1[jj], acc[jj]);

#pragma unroll
        for (int jj = 0; jj < 8; ++jj) {
            int kcol = grp * 128 + jj * 16 + l15;
#pragma unroll
            for (int v = 0; v < 8; ++v) {
                int q = qt * 16 + v + 8 * hi;   // local row 0..63
                float m = mask[(size_t)(qg0 + q) * DIM + kcol];
                s[q * 513 + kcol] = acc[jj][v] * SCALE + (1.0f - m) * -1e10f;
            }
        }
    }
    __syncthreads();

    // ---- softmax: 8 threads per row, bank-staggered, P stored f16 ----------
    {
        const int row = tid >> 3, seg = tid & 7;
        float* r = s + row * 513;
        float mx = -3.0e38f;
#pragma unroll 4
        for (int k = 0; k < 64; ++k) {
            int i = seg * 64 + ((k + seg * 8) & 63);
            mx = fmaxf(mx, r[i]);
        }
        red[row * 8 + seg] = mx;
        __syncthreads();
        float rmx = red[row * 8];
#pragma unroll
        for (int j = 1; j < 8; ++j) rmx = fmaxf(rmx, red[row * 8 + j]);
        float sum = 0.f;
#pragma unroll 4
        for (int k = 0; k < 64; ++k) {
            int i = seg * 64 + ((k + seg * 8) & 63);
            float e = __expf(r[i] - rmx);
            r[i] = e;
            sum += e;
        }
        red2[row * 8 + seg] = sum;
        __syncthreads();
        float tot = 0.f;
#pragma unroll
        for (int j = 0; j < 8; ++j) tot += red2[row * 8 + j];
        float inv = 1.0f / tot;
        _Float16* ph = Ph + row * 520;
#pragma unroll 4
        for (int k = 0; k < 64; ++k) {
            int i = seg * 64 + ((k + seg * 8) & 63);
            ph[i] = (_Float16)(r[i] * inv);
        }
    }
    __syncthreads();

    // ---- P @ V : ping-pong buffers, A (f16) from LDS, B from Vt ------------
    {
        const _Float16* prow = Ph + (qt * 16 + l15) * 520 + hi * 16;
        const _Float16* vp0  = Vt + ((size_t)blk * DIM + grp * 128 + l15) * DIM + hi * 16;
        auto ldA = [&](int kk) { return cat16(ld8h(prow + kk), ld8h(prow + kk + 8)); };
        auto ldB = [&](int kk, v16h* bb) {
#pragma unroll
            for (int jj = 0; jj < 8; ++jj) bb[jj] = ld16h(vp0 + jj * (16 * DIM) + kk);
        };

        v8f acc[8] = {};
        v16h a0, a1, b0[8], b1[8];
        a0 = ldA(0); ldB(0, b0);

#pragma unroll 1
        for (int kk = 0; kk < DIM - 64; kk += 64) {
            a1 = ldA(kk + 32); ldB(kk + 32, b1);
#pragma unroll
            for (int jj = 0; jj < 8; ++jj) acc[jj] = wmma16(a0, b0[jj], acc[jj]);
            a0 = ldA(kk + 64); ldB(kk + 64, b0);
#pragma unroll
            for (int jj = 0; jj < 8; ++jj) acc[jj] = wmma16(a1, b1[jj], acc[jj]);
        }
        a1 = ldA(DIM - 32); ldB(DIM - 32, b1);
#pragma unroll
        for (int jj = 0; jj < 8; ++jj) acc[jj] = wmma16(a0, b0[jj], acc[jj]);
#pragma unroll
        for (int jj = 0; jj < 8; ++jj) acc[jj] = wmma16(a1, b1[jj], acc[jj]);

        // stage out tiles into s (scores fully consumed)
#pragma unroll
        for (int jj = 0; jj < 8; ++jj) {
            int dcol = grp * 128 + jj * 16 + l15;
#pragma unroll
            for (int v = 0; v < 8; ++v)
                s[(qt * 16 + v + 8 * hi) * 513 + dcol] = acc[jj][v];
        }
    }
    __syncthreads();

    // ---- residual + LayerNorm stats: 8 threads per row ---------------------
    {
        const int row = tid >> 3, seg = tid & 7;
        const float* xr = X + (size_t)(qg0 + row) * DIM;
        float* r = s + row * 513;
        float sum = 0.f, sq = 0.f;
#pragma unroll 4
        for (int k = 0; k < 64; ++k) {
            int i = seg * 64 + ((k + seg * 8) & 63);
            float v = r[i] + xr[i];
            r[i] = v;
            sum += v; sq += v * v;
        }
        red[row * 8 + seg] = sum;
        red2[row * 8 + seg] = sq;
    }
    __syncthreads();
    if (tid < 64) {
        float sum = 0.f, sq = 0.f;
#pragma unroll
        for (int j = 0; j < 8; ++j) { sum += red[tid * 8 + j]; sq += red2[tid * 8 + j]; }
        float mean = sum * (1.0f / DIM);
        float var  = sq * (1.0f / DIM) - mean * mean;
        rMean[tid] = mean;
        rRstd[tid] = rsqrtf(fmaxf(var, 0.f) + LN_EPS);
    }
    __syncthreads();

    // ---- coalesced normalized store ----------------------------------------
    for (int i = tid; i < 64 * DIM; i += 512) {
        int q = i >> 9, c = i & 511;
        out[(size_t)(qg0 + q) * DIM + c] = (s[q * 513 + c] - rMean[q]) * rRstd[q];
    }
}

// ---------------------------------------------------------------------------
extern "C" void kernel_launch(void* const* d_in, const int* in_sizes, int n_in,
                              void* d_out, int out_size, void* d_ws, size_t ws_size,
                              hipStream_t stream)
{
    (void)in_sizes; (void)n_in; (void)out_size; (void)ws_size;
    const float* X    = (const float*)d_in[0];
    const float* mask = (const float*)d_in[1];
    const float* dw1  = (const float*)d_in[2];
    const float* dw2  = (const float*)d_in[3];
    const float* dw3  = (const float*)d_in[4];
    const float* db1  = (const float*)d_in[5];
    const float* db2  = (const float*)d_in[6];
    const float* db3  = (const float*)d_in[7];

    _Float16* h  = (_Float16*)d_ws;
    _Float16* Qh = h + QH_OFF;
    _Float16* Kh = h + KH_OFF;
    _Float16* Vt = h + VT_OFF;
    _Float16* Wt = h + WT_OFF;
    _Float16* Xh = h + XH_OFF;

    // 1) weights -> f16 transposed ; X -> f16
    prep_weights<<<(3 * DIM * DIM + 255) / 256, 256, 0, stream>>>(dw1, dw2, dw3, Wt);
    prep_xh<<<8192, 256, 0, stream>>>(X, Xh);
    // 2) QKV projections: 3 * 2048 * 8 strips / 8 waves per block
    qkv_gemm<<<6144, 256, 0, stream>>>(Xh, db1, db2, db3, Wt, Qh, Kh, Vt);
    // 3) fused attention + layernorm: 64 blocks * 8 q-chunks
    attn_fused<<<512, 512, 0, stream>>>(X, mask, Qh, Kh, Vt, (float*)d_out);
}
